// PlatonicConv_35527969472529
// MI455X (gfx1250) — matrix operand order repaired and checked by reading, not compiled, not run on hardware
//
#include <hip/hip_runtime.h>
#include <hip/hip_bf16.h>
#include <math.h>

// Problem constants (from reference)
#define N_NODES 20000
#define E_EDGES 256000
#define G_ 12
#define H_ 2
#define D_ 16
#define GH_ (G_ * H_)        // 24
#define EMB_ (GH_ * D_)      // 384
#define IN_ 384
#define OUT_ 384
#define SD_ 3
#define EPS_ 1e-6f
#define SCALE_ 0.25f         // 16^-0.5
#define MTILES_ (N_NODES / 16)   // 1250

typedef _Float16 h8   __attribute__((ext_vector_type(8)));
typedef _Float16 v16h __attribute__((ext_vector_type(16)));
typedef float    v8f  __attribute__((ext_vector_type(8)));
typedef int      v4i  __attribute__((ext_vector_type(4)));

// CDNA5 async global->LDS path (ASYNCcnt); falls back to load+ds_store staging.
#if __has_builtin(__builtin_amdgcn_global_load_async_to_lds_b128) && \
    __has_builtin(__builtin_amdgcn_s_wait_asynccnt)
#define HAVE_ASYNC_LDS 1
#else
#define HAVE_ASYNC_LDS 0
#endif

__device__ __forceinline__ void stage_16B(const _Float16* g, _Float16* l) {
#if HAVE_ASYNC_LDS
    // Prototype: (v4i addrspace(1)*, v4i addrspace(3)*, imm offset, imm cpol)
    __builtin_amdgcn_global_load_async_to_lds_b128(
        (__attribute__((address_space(1))) v4i*)g,
        (__attribute__((address_space(3))) v4i*)l, 0, 0);
#else
    *(float4*)l = *(const float4*)g;
#endif
}

__device__ __forceinline__ void stage_wait_barrier() {
#if HAVE_ASYNC_LDS
    __builtin_amdgcn_s_wait_asynccnt(0);
#endif
    __syncthreads();
}

// ---------------- elementwise converts ----------------

__global__ void cvt_f32_to_f16_kernel(const float* __restrict__ src,
                                      _Float16* __restrict__ dst, int n) {
    int i = blockIdx.x * blockDim.x + threadIdx.x;
    if (i < n) dst[i] = (_Float16)src[i];
}

// W: rows x cols (row-major, f32) -> WT: cols x rows (row-major, f16)
__global__ void transpose_f32_to_f16_kernel(const float* __restrict__ W,
                                            _Float16* __restrict__ WT,
                                            int rows, int cols) {
    int i = blockIdx.x * blockDim.x + threadIdx.x;
    if (i < rows * cols) {
        int r = i / cols, c = i % cols;
        WT[(size_t)c * rows + r] = (_Float16)W[i];
    }
}

// ---------------- WMMA operand loaders ----------------

// A operand: 16x32 f16 tile (ISA 7.12.2 layout). M = lane&15;
// K = (lane>>4)*8 + e (e<8) and 16 + (lane>>4)*8 + (e-8) (e>=8).
__device__ __forceinline__ v16h load_A_tile(const _Float16* __restrict__ rowBase,
                                            int k0, int lane) {
    int kg = (lane >> 4) * 8;
    h8 lo = *(const h8*)(rowBase + k0 + kg);
    h8 hi = *(const h8*)(rowBase + k0 + 16 + kg);
    v16h a;
#pragma unroll
    for (int j = 0; j < 8; ++j) { a[j] = lo[j]; a[j + 8] = hi[j]; }
    return a;
}

// B operand from a K-major row (pre-transposed weights): N = lane&15;
// K = (lane>>4)*8 + e (e<8) and 16 + (lane>>4)*8 + (e-8) (e>=8).
__device__ __forceinline__ v16h load_B_tile(const _Float16* __restrict__ colBase,
                                            int k0, int lane) {
    int kg = (lane >> 4) * 8;
    h8 lo = *(const h8*)(colBase + k0 + kg);
    h8 hi = *(const h8*)(colBase + k0 + 16 + kg);
    v16h b;
#pragma unroll
    for (int j = 0; j < 8; ++j) { b[j] = lo[j]; b[j + 8] = hi[j]; }
    return b;
}

// ---------------- fused Q/V projection GEMM ----------------
// 256 threads = 8 waves = 4 m-tiles x 2 n-tiles. Weight panels (32 out-cols x
// 384 K) staged in LDS once per block via async-to-LDS, shared by all waves.
__global__ __launch_bounds__(256) void gemm_qv_wmma_kernel(
    const _Float16* __restrict__ A,
    const _Float16* __restrict__ BqT,
    const _Float16* __restrict__ BvT,
    const float* __restrict__ bq, const float* __restrict__ bv,
    float* __restrict__ Q, float* __restrict__ V) {
    __shared__ __align__(16) _Float16 lBq[32 * IN_];
    __shared__ __align__(16) _Float16 lBv[32 * IN_];

    const int tid   = threadIdx.x;
    const int nBase = blockIdx.y * 32;                 // 2 n-tiles per block
    const _Float16* gBq = BqT + (size_t)nBase * IN_;   // contiguous 24KB panel
    const _Float16* gBv = BvT + (size_t)nBase * IN_;
    const int CH = 32 * IN_ / 8;                       // 1536 16B chunks
    for (int i = tid; i < CH; i += 256) {
        stage_16B(gBq + i * 8, lBq + i * 8);
        stage_16B(gBv + i * 8, lBv + i * 8);
    }
    stage_wait_barrier();

    const int wave   = tid >> 5;
    const int lane   = tid & 31;
    const int mTile  = blockIdx.x * 4 + (wave >> 1);   // wave-uniform
    const int nLocal = wave & 1;
    if (mTile < MTILES_) {                             // uniform per wave
        const _Float16* arow  = A + (size_t)(mTile * 16 + (lane & 15)) * IN_;
        const _Float16* bqrow = lBq + (nLocal * 16 + (lane & 15)) * IN_;
        const _Float16* bvrow = lBv + (nLocal * 16 + (lane & 15)) * IN_;

        v8f cq = {0.f, 0.f, 0.f, 0.f, 0.f, 0.f, 0.f, 0.f};
        v8f cv = {0.f, 0.f, 0.f, 0.f, 0.f, 0.f, 0.f, 0.f};
#pragma unroll 4
        for (int k0 = 0; k0 < IN_; k0 += 32) {
            if (k0 + 32 < IN_)
                __builtin_prefetch(arow + k0 + 32, 0, 0);  // global_prefetch_b8
            v16h a  = load_A_tile(arow, k0, lane);
            v16h b0 = load_B_tile(bqrow, k0, lane);        // ds_load path
            cq = __builtin_amdgcn_wmma_f32_16x16x32_f16(false, a, false, b0,
                                                        (short)0, cq, false, false);
            v16h b1 = load_B_tile(bvrow, k0, lane);
            cv = __builtin_amdgcn_wmma_f32_16x16x32_f16(false, a, false, b1,
                                                        (short)0, cv, false, false);
        }
        const int col = nBase + nLocal * 16 + (lane & 15);
        const float biasq = bq[col];
        const float biasv = bv[col];
#pragma unroll
        for (int i = 0; i < 8; ++i) {
            int row = mTile * 16 + i + 8 * (lane >> 4);
            Q[(size_t)row * EMB_ + col] = cq[i] + biasq;
            V[(size_t)row * EMB_ + col] = cv[i] + biasv;
        }
    }
}

// ---------------- output GEMM ----------------
// C = A @ Wo + bo. Same 8-wave / LDS-staged structure, single weight panel.
__global__ __launch_bounds__(256) void gemm_out_wmma_kernel(
    const _Float16* __restrict__ A, const _Float16* __restrict__ BT,
    const float* __restrict__ bias, float* __restrict__ C) {
    __shared__ __align__(16) _Float16 lB[32 * EMB_];

    const int tid   = threadIdx.x;
    const int nBase = blockIdx.y * 32;
    const _Float16* gB = BT + (size_t)nBase * EMB_;
    const int CH = 32 * EMB_ / 8;
    for (int i = tid; i < CH; i += 256)
        stage_16B(gB + i * 8, lB + i * 8);
    stage_wait_barrier();

    const int wave   = tid >> 5;
    const int lane   = tid & 31;
    const int mTile  = blockIdx.x * 4 + (wave >> 1);
    const int nLocal = wave & 1;
    if (mTile < MTILES_) {
        const _Float16* arow = A + (size_t)(mTile * 16 + (lane & 15)) * EMB_;
        const _Float16* brow = lB + (nLocal * 16 + (lane & 15)) * EMB_;

        v8f c = {0.f, 0.f, 0.f, 0.f, 0.f, 0.f, 0.f, 0.f};
#pragma unroll 4
        for (int k0 = 0; k0 < EMB_; k0 += 32) {
            if (k0 + 32 < EMB_)
                __builtin_prefetch(arow + k0 + 32, 0, 0);
            v16h a = load_A_tile(arow, k0, lane);
            v16h b = load_B_tile(brow, k0, lane);
            c = __builtin_amdgcn_wmma_f32_16x16x32_f16(false, a, false, b,
                                                       (short)0, c, false, false);
        }
        const int col = nBase + nLocal * 16 + (lane & 15);
        const float bi = bias[col];
#pragma unroll
        for (int i = 0; i < 8; ++i) {
            int row = mTile * 16 + i + 8 * (lane >> 4);
            C[(size_t)row * OUT_ + col] = c[i] + bi;
        }
    }
}

// ---------------- RMSNorm + RoPE (q in place) and constant-k ----------------

__global__ void ropenorm_kernel(float* __restrict__ q,          // N x EMB (in place)
                                float* __restrict__ kf,         // N x H x D
                                const float* __restrict__ pos,  // N x 3
                                const float* __restrict__ freqs,// H x 8 x 3
                                const float* __restrict__ qn_w, // D
                                const float* __restrict__ kn_w) // D
{
    int idx = blockIdx.x * blockDim.x + threadIdx.x;
    if (idx >= N_NODES * GH_) return;
    int n  = idx / GH_;
    int gh = idx % GH_;
    int g  = gh / H_;
    int h  = gh % H_;

    float p0 = pos[n * SD_ + 0], p1 = pos[n * SD_ + 1], p2 = pos[n * SD_ + 2];
    float cs[8], sn[8];
#pragma unroll
    for (int d2 = 0; d2 < 8; ++d2) {
        const float* fr = freqs + ((size_t)h * 8 + d2) * SD_;
        float ang = p0 * fr[0] + p1 * fr[1] + p2 * fr[2];
        cs[d2] = __cosf(ang);
        sn[d2] = __sinf(ang);
    }

    float* qp = q + (size_t)n * EMB_ + gh * D_;
    float t[16];
    float ss = 0.f;
#pragma unroll
    for (int d = 0; d < 16; ++d) { t[d] = qp[d]; ss += t[d] * t[d]; }
    float inv = rsqrtf(ss * (1.f / 16.f) + EPS_);
#pragma unroll
    for (int d = 0; d < 16; ++d) t[d] *= inv * qn_w[d];
#pragma unroll
    for (int i = 0; i < 8; ++i) {
        float t1 = t[2 * i], t2 = t[2 * i + 1];
        qp[2 * i]     = t1 * cs[i] - t2 * sn[i];
        qp[2 * i + 1] = t1 * sn[i] + t2 * cs[i];
    }

    if (g == 0) {   // k = rope(rmsnorm(ones, kn_w)) — depends only on (n, h)
        float invk = rsqrtf(1.f + EPS_);
        float* kp = kf + ((size_t)n * H_ + h) * D_;
#pragma unroll
        for (int i = 0; i < 8; ++i) {
            float k1 = invk * kn_w[2 * i];
            float k2 = invk * kn_w[2 * i + 1];
            kp[2 * i]     = k1 * cs[i] - k2 * sn[i];
            kp[2 * i + 1] = k1 * sn[i] + k2 * cs[i];
        }
    }
}

// ---------------- segment softmax + aggregation ----------------

__global__ void init_kernel(float* __restrict__ m, float* __restrict__ s,
                            float* __restrict__ agg) {
    int i = blockIdx.x * blockDim.x + threadIdx.x;
    if (i < N_NODES * GH_) { m[i] = -INFINITY; s[i] = 0.f; }
    if (i < N_NODES * EMB_) agg[i] = 0.f;
}

__device__ __forceinline__ void atomicMaxF(float* addr, float val) {
    // Monotone in both halves of the float encoding; init must be -inf.
    if (val >= 0.f) atomicMax((int*)addr, __float_as_int(val));
    else            atomicMin((unsigned int*)addr, __float_as_uint(val));
}

__global__ void score_kernel(const float* __restrict__ q,   // N x EMB (post rope)
                             const float* __restrict__ kf,  // N x H x D
                             const int* __restrict__ src,
                             const int* __restrict__ dst,
                             float* __restrict__ scores,    // E x GH
                             float* __restrict__ m) {       // N x GH
    int idx = blockIdx.x * blockDim.x + threadIdx.x;
    if (idx >= E_EDGES * GH_) return;
    int e  = idx / GH_;
    int gh = idx % GH_;
    int h  = gh % H_;
    int sN = src[e], dN = dst[e];
    const float4* qp = (const float4*)(q + (size_t)sN * EMB_ + gh * D_);
    const float4* kp = (const float4*)(kf + ((size_t)dN * H_ + h) * D_);
    float acc = 0.f;
#pragma unroll
    for (int j = 0; j < 4; ++j) {
        float4 a = qp[j], b = kp[j];
        acc += a.x * b.x + a.y * b.y + a.z * b.z + a.w * b.w;
    }
    acc *= SCALE_;
    scores[idx] = acc;
    atomicMaxF(&m[(size_t)sN * GH_ + gh], acc);
}

__global__ void expsum_kernel(const int* __restrict__ src,
                              const float* __restrict__ m,
                              float* __restrict__ scores,   // in place -> exp()
                              float* __restrict__ s) {
    int idx = blockIdx.x * blockDim.x + threadIdx.x;
    if (idx >= E_EDGES * GH_) return;
    int e  = idx / GH_;
    int gh = idx % GH_;
    int sN = src[e];
    float ev = __expf(scores[idx] - m[(size_t)sN * GH_ + gh]);
    scores[idx] = ev;
    atomicAdd(&s[(size_t)sN * GH_ + gh], ev);
}

__global__ void aggregate_kernel(const int* __restrict__ src,
                                 const int* __restrict__ dst,
                                 const float* __restrict__ scores, // exp values
                                 const float* __restrict__ s,
                                 const float* __restrict__ v,      // N x EMB
                                 float* __restrict__ agg) {        // N x EMB
    int idx = blockIdx.x * blockDim.x + threadIdx.x;
    if (idx >= E_EDGES * GH_) return;
    int e  = idx / GH_;
    int gh = idx % GH_;
    int sN = src[e], dN = dst[e];
    float a = scores[idx] / (s[(size_t)sN * GH_ + gh] + 1e-16f);
    const float* vp = v + (size_t)dN * EMB_ + gh * D_;
    float* ap = agg + (size_t)sN * EMB_ + gh * D_;
#pragma unroll
    for (int d = 0; d < 16; ++d) atomicAdd(&ap[d], a * vp[d]);
}

// ---------------- launch ----------------

extern "C" void kernel_launch(void* const* d_in, const int* in_sizes, int n_in,
                              void* d_out, int out_size, void* d_ws, size_t ws_size,
                              hipStream_t stream) {
    (void)in_sizes; (void)n_in; (void)out_size; (void)ws_size;
    const float* x     = (const float*)d_in[0];
    const float* pos   = (const float*)d_in[1];
    const int*   src   = (const int*)d_in[2];
    const int*   dst   = (const int*)d_in[3];
    const float* Wq    = (const float*)d_in[4];
    const float* bq    = (const float*)d_in[5];
    const float* Wv    = (const float*)d_in[6];
    const float* bv    = (const float*)d_in[7];
    const float* Wo    = (const float*)d_in[8];
    const float* bo    = (const float*)d_in[9];
    const float* freqs = (const float*)d_in[10];
    const float* qn_w  = (const float*)d_in[11];
    const float* kn_w  = (const float*)d_in[12];
    float* out = (float*)d_out;

    char* ws = (char*)d_ws;
    size_t off = 0;
    auto alloc = [&](size_t bytes) {
        size_t o = off;
        off += (bytes + 255) & ~(size_t)255;
        return o;
    };
    _Float16* xh   = (_Float16*)(ws + alloc((size_t)N_NODES * IN_ * 2));
    _Float16* wqT  = (_Float16*)(ws + alloc((size_t)EMB_ * IN_ * 2));
    _Float16* wvT  = (_Float16*)(ws + alloc((size_t)EMB_ * IN_ * 2));
    _Float16* woT  = (_Float16*)(ws + alloc((size_t)OUT_ * EMB_ * 2));
    float* qbuf    = (float*)(ws + alloc((size_t)N_NODES * EMB_ * 4));
    float* vbuf    = (float*)(ws + alloc((size_t)N_NODES * EMB_ * 4));
    float* kf      = (float*)(ws + alloc((size_t)N_NODES * H_ * D_ * 4));
    float* scores  = (float*)(ws + alloc((size_t)E_EDGES * GH_ * 4));
    float* mbuf    = (float*)(ws + alloc((size_t)N_NODES * GH_ * 4));
    float* sbuf    = (float*)(ws + alloc((size_t)N_NODES * GH_ * 4));
    float* agg     = (float*)(ws + alloc((size_t)N_NODES * EMB_ * 4));
    _Float16* aggh = (_Float16*)(ws + alloc((size_t)N_NODES * EMB_ * 2));

    const int TB = 256;
    // Convert/transpose operands to f16
    cvt_f32_to_f16_kernel<<<(N_NODES * IN_ + TB - 1) / TB, TB, 0, stream>>>(x, xh, N_NODES * IN_);
    transpose_f32_to_f16_kernel<<<(IN_ * EMB_ + TB - 1) / TB, TB, 0, stream>>>(Wq, wqT, IN_, EMB_);
    transpose_f32_to_f16_kernel<<<(IN_ * EMB_ + TB - 1) / TB, TB, 0, stream>>>(Wv, wvT, IN_, EMB_);
    transpose_f32_to_f16_kernel<<<(EMB_ * OUT_ + TB - 1) / TB, TB, 0, stream>>>(Wo, woT, EMB_, OUT_);

    // Fused Q/V projection GEMM (WMMA, LDS-staged weight panels)
    dim3 qvgrid((MTILES_ + 3) / 4, EMB_ / 32);
    gemm_qv_wmma_kernel<<<qvgrid, 256, 0, stream>>>(xh, wqT, wvT, bq, bv, qbuf, vbuf);

    // RMSNorm + RoPE on q (in place) and constant k
    ropenorm_kernel<<<(N_NODES * GH_ + TB - 1) / TB, TB, 0, stream>>>(
        qbuf, kf, pos, freqs, qn_w, kn_w);

    // Segment softmax state init
    init_kernel<<<(N_NODES * EMB_ + TB - 1) / TB, TB, 0, stream>>>(mbuf, sbuf, agg);

    int egrid = (E_EDGES * GH_ + TB - 1) / TB;
    score_kernel<<<egrid, TB, 0, stream>>>(qbuf, kf, src, dst, scores, mbuf);
    expsum_kernel<<<egrid, TB, 0, stream>>>(src, mbuf, scores, sbuf);
    aggregate_kernel<<<egrid, TB, 0, stream>>>(src, dst, scores, sbuf, vbuf, agg);

    // Output GEMM (WMMA)
    cvt_f32_to_f16_kernel<<<(N_NODES * EMB_ + TB - 1) / TB, TB, 0, stream>>>(agg, aggh, N_NODES * EMB_);
    dim3 ogrid((MTILES_ + 3) / 4, OUT_ / 32);
    gemm_out_wmma_kernel<<<ogrid, 256, 0, stream>>>(aggh, woT, bo, out);
}
